// TransConv_40827959116204
// MI455X (gfx1250) — compile-verified, matrix operand order checked by reading, not run on hardware
//
#include <hip/hip_runtime.h>
#include <cstdint>

// ---------------------------------------------------------------------------
// Problem constants (from reference):  B=16,T=12,N=883,D=64,H=128,NH=4
#define L_TOK   169536          // B*T*N tokens; 169536 = 64 * 2649
#define D_INP   64
#define HIDC    128
#define NHEAD   4
#define TGT     64              // tokens per workgroup tile
#define NTILE   (L_TOK / TGT)   // 2649
#define NTHR    256             // 8 waves (wave32)
#define LN_EPS  1e-5f
#define ATT_EPS 1e-8f

typedef __attribute__((ext_vector_type(16))) _Float16 v16h;
typedef __attribute__((ext_vector_type(8)))  float    v8f;

union FragU { v16h h; uint4 u[2]; };

__device__ __forceinline__ v8f v8zero() {
  v8f z = {0.f,0.f,0.f,0.f,0.f,0.f,0.f,0.f};
  return z;
}

// Load one 16(M)x32(K) f16 A-fragment.  Element (m,k) lives at base[m*ld + k].
// CDNA5 16-bit A layout: lane m = lane&15; K chunks {k0..k0+7} and
// {k0+16..k0+23} with k0 = (lane<16 ? 0 : 8)  ->  two 16B loads per lane.
// The B-fragment (32Kx16N) uses the mirrored layout over N, so the same
// loader is used with a pre-transposed  [N][K]  weight buffer.
__device__ __forceinline__ v16h load_frag16(const _Float16* base, int ld, int lane) {
  const int r  = lane & 15;
  const int k0 = (lane & 16) ? 8 : 0;
  const _Float16* p = base + r * ld + k0;
  FragU f;
  f.u[0] = *(const uint4*)(p);
  f.u[1] = *(const uint4*)(p + 16);
  return f.h;
}

__device__ __forceinline__ v8f wmma_f16(v16h a, v16h b, v8f c) {
  // v_wmma_f32_16x16x32_f16  (doc-verified codegen path on gfx1250)
  return __builtin_amdgcn_wmma_f32_16x16x32_f16(false, a, false, b, (short)0, c,
                                                false, false);
}

// One K-step-pipelined 4xM-tile GEMM slice: all fragment loads are issued
// before the 4 WMMAs so a single counter wait covers the whole K-step.
template<int KSTEPS>
__device__ __forceinline__ void gemm_4m(const _Float16* __restrict__ A, int lda,
                                        const _Float16* __restrict__ B, int ldb,
                                        v8f* acc, int lane) {
#pragma unroll
  for (int ks = 0; ks < KSTEPS; ++ks) {
    v16h b  = load_frag16(B + ks * 32, ldb, lane);
    v16h a0 = load_frag16(A + (0 * 16) * lda + ks * 32, lda, lane);
    v16h a1 = load_frag16(A + (1 * 16) * lda + ks * 32, lda, lane);
    v16h a2 = load_frag16(A + (2 * 16) * lda + ks * 32, lda, lane);
    v16h a3 = load_frag16(A + (3 * 16) * lda + ks * 32, lda, lane);
    acc[0] = wmma_f16(a0, b, acc[0]);
    acc[1] = wmma_f16(a1, b, acc[1]);
    acc[2] = wmma_f16(a2, b, acc[2]);
    acc[3] = wmma_f16(a3, b, acc[3]);
  }
}

__device__ __forceinline__ void atomAddLds(float* p, float v) {
  __hip_atomic_fetch_add(p, v, __ATOMIC_RELAXED, __HIP_MEMORY_SCOPE_WORKGROUP);
}
__device__ __forceinline__ void atomAddGlb(float* p, float v) {
  __hip_atomic_fetch_add(p, v, __ATOMIC_RELAXED, __HIP_MEMORY_SCOPE_AGENT);
}

// ---------------------------------------------------------------------------
// prep kernels
__global__ void k_zero(float* __restrict__ p, int n) {
  int i = blockIdx.x * NTHR + threadIdx.x;
  if (i < n) p[i] = 0.f;
}

// dst[c*rows + r] = (f16) src[r*cols + c]   (weights -> transposed f16)
__global__ void k_cast_t(const float* __restrict__ src, _Float16* __restrict__ dst,
                         int rows, int cols) {
  int i = blockIdx.x * NTHR + threadIdx.x;
  if (i < rows * cols) {
    int r = i / cols, c = i - r * cols;
    dst[c * rows + r] = (_Float16)src[i];
  }
}

// ---------------------------------------------------------------------------
// Pass 1: fc0 GEMM + LN + ReLU -> h ;  per-head K,V GEMMs ; kvs/ks/vs reduction
__global__ __launch_bounds__(NTHR, 1)
void k_front(const float* __restrict__ x,
             const _Float16* __restrict__ fc0T, const float* __restrict__ fc0_b,
             const float* __restrict__ ln0_g,  const float* __restrict__ ln0_b,
             const _Float16* __restrict__ wkT, const float* __restrict__ wk_b,
             const _Float16* __restrict__ wvT, const float* __restrict__ wv_b,
             _Float16* __restrict__ hbuf,
             float* __restrict__ kvs, float* __restrict__ ksum,
             float* __restrict__ vsum)
{
  __shared__ _Float16 sX [TGT * D_INP];    //  8 KB  x tile (f16)
  __shared__ _Float16 sH [TGT * HIDC];     // 16 KB  h tile (post LN+ReLU, f16)
  __shared__ _Float16 sKT[HIDC * TGT];     // 16 KB  k^T (feature-major, f16)
  __shared__ _Float16 sVT[HIDC * TGT];     // 16 KB  v^T
  __shared__ float red0[TGT], red1[TGT], mu_s[TGT], rs_s[TGT];

  const int tid  = threadIdx.x;
  const int lane = tid & 31;
  const int wv   = tid >> 5;               // wave id 0..7
  const int tok0 = blockIdx.x * TGT;
  const int col  = wv * 16 + (lane & 15);  // output column (N) for GEMM phases
  const int rhi  = (lane & 16) ? 8 : 0;    // C-layout: VGPR r holds M=r / r+8

  { // stage x tile as f16 (vectorized: float4 in, uint2 out)
    const float4* xs = (const float4*)(x + (size_t)tok0 * D_INP);
    for (int j = tid; j < TGT * D_INP / 4; j += NTHR) {
      float4 v = xs[j];
      union { _Float16 h[4]; uint2 u; } pk;
      pk.h[0] = (_Float16)v.x; pk.h[1] = (_Float16)v.y;
      pk.h[2] = (_Float16)v.z; pk.h[3] = (_Float16)v.w;
      *(uint2*)(sX + 4 * j) = pk.u;
    }
  }
  if (tid < TGT) { red0[tid] = 0.f; red1[tid] = 0.f; }
  __syncthreads();

  // ---- GEMM1: h = x @ fc0_w  (M=64,K=64,N=128); wave wv owns n-tile wv
  v8f acc[4];
#pragma unroll
  for (int mt = 0; mt < 4; ++mt) acc[mt] = v8zero();
  gemm_4m<2>(sX, D_INP, fc0T + (wv * 16) * D_INP, D_INP, acc, lane);
  {
    const float fb = fc0_b[col];
#pragma unroll
    for (int mt = 0; mt < 4; ++mt)
#pragma unroll
      for (int r = 0; r < 8; ++r) {
        float vv = acc[mt][r] + fb;
        acc[mt][r] = vv;
        int row = mt * 16 + r + rhi;
        atomAddLds(&red0[row], vv);
        atomAddLds(&red1[row], vv * vv);
      }
  }
  __syncthreads();
  if (tid < TGT) {
    float mu  = red0[tid] * (1.f / HIDC);
    float var = red1[tid] * (1.f / HIDC) - mu * mu;
    mu_s[tid] = mu;
    rs_s[tid] = rsqrtf(var + LN_EPS);
  }
  __syncthreads();
  {
    const float g = ln0_g[col], bb = ln0_b[col];
#pragma unroll
    for (int mt = 0; mt < 4; ++mt)
#pragma unroll
      for (int r = 0; r < 8; ++r) {
        int row = mt * 16 + r + rhi;
        float vv = (acc[mt][r] - mu_s[row]) * rs_s[row] * g + bb;
        vv = fmaxf(vv, 0.f);
        sH[row * HIDC + col] = (_Float16)vv;
      }
  }
  __syncthreads();

  { // h tile -> hbuf, coalesced b128 (replaces 8192 scattered b16 stores)
    const uint4* s = (const uint4*)sH;
    uint4* d = (uint4*)(hbuf + (size_t)tok0 * HIDC);
    for (int i = tid; i < TGT * HIDC / 8; i += NTHR) d[i] = s[i];
  }

  // ---- per head:  K,V GEMMs (M=64,K=128,N=128)  then  kvs += K^T V
#pragma unroll 1
  for (int hh = 0; hh < NHEAD; ++hh) {
    if (tid < TGT) red0[tid] = 0.f;
    __syncthreads();                       // also guards sKT/sVT reuse

    // K = h @ wk + b ; row-L2-normalize ; transpose into sKT
#pragma unroll
    for (int mt = 0; mt < 4; ++mt) acc[mt] = v8zero();
    gemm_4m<4>(sH, HIDC, wkT + (hh * HIDC + wv * 16) * HIDC, HIDC, acc, lane);
    {
      const float kb = wk_b[hh * HIDC + col];
#pragma unroll
      for (int mt = 0; mt < 4; ++mt)
#pragma unroll
        for (int r = 0; r < 8; ++r) {
          float vv = acc[mt][r] + kb;
          acc[mt][r] = vv;
          atomAddLds(&red0[mt * 16 + r + rhi], vv * vv);
        }
    }
    __syncthreads();
    if (tid < TGT) rs_s[tid] = 1.f / (sqrtf(red0[tid]) + ATT_EPS);
    __syncthreads();
    {
      float kpart = 0.f;
#pragma unroll
      for (int mt = 0; mt < 4; ++mt)
#pragma unroll
        for (int r = 0; r < 8; ++r) {
          int row = mt * 16 + r + rhi;
          float vv = acc[mt][r] * rs_s[row];
          sKT[col * TGT + row] = (_Float16)vv;
          kpart += vv;
        }
      atomAddGlb(&ksum[hh * HIDC + col], kpart);
    }

    // V = h @ wv + b ; transpose into sVT
#pragma unroll
    for (int mt = 0; mt < 4; ++mt) acc[mt] = v8zero();
    gemm_4m<4>(sH, HIDC, wvT + (hh * HIDC + wv * 16) * HIDC, HIDC, acc, lane);
    {
      const float vb = wv_b[hh * HIDC + col];
      float vpart = 0.f;
#pragma unroll
      for (int mt = 0; mt < 4; ++mt)
#pragma unroll
        for (int r = 0; r < 8; ++r) {
          int row = mt * 16 + r + rhi;
          float vvv = acc[mt][r] + vb;
          sVT[col * TGT + row] = (_Float16)vvv;
          vpart += vvv;
        }
      atomAddGlb(&vsum[hh * HIDC + col], vpart);
    }
    __syncthreads();                       // sKT / sVT ready

    // kvs_head += K^T @ V  (M=128 feat, K=64 tok, N=128 d); wave wv: m-tile wv
    // pipelined: A frag + all 8 B frags loaded before the 8 WMMAs per K-step
    v8f ka[8];
#pragma unroll
    for (int dt = 0; dt < 8; ++dt) ka[dt] = v8zero();
#pragma unroll
    for (int ks = 0; ks < 2; ++ks) {
      v16h a = load_frag16(sKT + (wv * 16) * TGT + ks * 32, TGT, lane);
      v16h b0 = load_frag16(sVT + (0 * 16) * TGT + ks * 32, TGT, lane);
      v16h b1 = load_frag16(sVT + (1 * 16) * TGT + ks * 32, TGT, lane);
      v16h b2 = load_frag16(sVT + (2 * 16) * TGT + ks * 32, TGT, lane);
      v16h b3 = load_frag16(sVT + (3 * 16) * TGT + ks * 32, TGT, lane);
      v16h b4 = load_frag16(sVT + (4 * 16) * TGT + ks * 32, TGT, lane);
      v16h b5 = load_frag16(sVT + (5 * 16) * TGT + ks * 32, TGT, lane);
      v16h b6 = load_frag16(sVT + (6 * 16) * TGT + ks * 32, TGT, lane);
      v16h b7 = load_frag16(sVT + (7 * 16) * TGT + ks * 32, TGT, lane);
      ka[0] = wmma_f16(a, b0, ka[0]);
      ka[1] = wmma_f16(a, b1, ka[1]);
      ka[2] = wmma_f16(a, b2, ka[2]);
      ka[3] = wmma_f16(a, b3, ka[3]);
      ka[4] = wmma_f16(a, b4, ka[4]);
      ka[5] = wmma_f16(a, b5, ka[5]);
      ka[6] = wmma_f16(a, b6, ka[6]);
      ka[7] = wmma_f16(a, b7, ka[7]);
    }
    {
      float* kvh = kvs + hh * HIDC * HIDC;
#pragma unroll
      for (int dt = 0; dt < 8; ++dt)
#pragma unroll
        for (int r = 0; r < 8; ++r) {
          int m = wv * 16 + r + rhi;
          int d = dt * 16 + (lane & 15);
          atomAddGlb(&kvh[m * HIDC + d], ka[dt][r]);
        }
    }
    // next iteration's leading __syncthreads() guards sKT/sVT reuse
  }
}

// ---------------------------------------------------------------------------
// Pass 1.5: kvs f32 -> f16 transposed-per-head ([d][m]) for B fragments; v_mean
__global__ void k_mid(const float* __restrict__ kvs, _Float16* __restrict__ kvsT,
                      const float* __restrict__ vsum, float* __restrict__ vmean)
{
  int i = blockIdx.x * NTHR + threadIdx.x;
  if (i < NHEAD * HIDC * HIDC) {
    int hh = i >> 14;
    int rem = i & 16383;
    int m = rem >> 7, d = rem & 127;
    kvsT[hh * HIDC * HIDC + d * HIDC + m] = (_Float16)kvs[i];
  }
  if (i < NHEAD * HIDC) vmean[i] = vsum[i] * (1.f / (float)L_TOK);
}

// ---------------------------------------------------------------------------
// Pass 2: recompute q, normalize, denom, num = q @ kvs, head-mean, residual, LN
__global__ __launch_bounds__(NTHR, 1)
void k_attn(const _Float16* __restrict__ hbuf,
            const _Float16* __restrict__ wqT, const float* __restrict__ wq_b,
            const _Float16* __restrict__ kvsT, const float* __restrict__ ksum,
            const float* __restrict__ vmean,
            const float* __restrict__ ln1_g, const float* __restrict__ ln1_b,
            float* __restrict__ out)
{
  __shared__ _Float16 sH[TGT * HIDC];                      // 16 KB
  __shared__ __align__(16) unsigned char smem[TGT * HIDC * 4]; // 32 KB (sQ|sOut)
  _Float16* sQ   = (_Float16*)smem;                        // f16 q tile (16 KB)
  float*    sOut = (float*)smem;                           // f32 out tile (32 KB)
  __shared__ float red0[TGT], red1[TGT], rs_s[TGT], rden[TGT], mu_s[TGT];
  __shared__ float sKs[HIDC], sVm[HIDC];

  const int tid  = threadIdx.x;
  const int lane = tid & 31;
  const int wv   = tid >> 5;
  const int tok0 = blockIdx.x * TGT;
  const int col  = wv * 16 + (lane & 15);
  const int rhi  = (lane & 16) ? 8 : 0;

  { // stage h tile (coalesced b128)
    const uint4* src = (const uint4*)(hbuf + (size_t)tok0 * HIDC);
    uint4* dst = (uint4*)sH;
    for (int i = tid; i < TGT * HIDC / 8; i += NTHR) dst[i] = src[i];
  }
  v8f oacc[4];
#pragma unroll
  for (int mt = 0; mt < 4; ++mt) oacc[mt] = v8zero();

#pragma unroll 1
  for (int hh = 0; hh < NHEAD; ++hh) {
    __syncthreads();                       // previous head fully consumed
    if (tid < TGT)  { red0[tid] = 0.f; red1[tid] = 0.f; }
    if (tid < HIDC) { sKs[tid] = ksum[hh * HIDC + tid];
                      sVm[tid] = vmean[hh * HIDC + tid]; }
    __syncthreads();

    // Q = h @ wq + b
    v8f acc[4];
#pragma unroll
    for (int mt = 0; mt < 4; ++mt) acc[mt] = v8zero();
    gemm_4m<4>(sH, HIDC, wqT + (hh * HIDC + wv * 16) * HIDC, HIDC, acc, lane);
    {
      const float qb = wq_b[hh * HIDC + col];
#pragma unroll
      for (int mt = 0; mt < 4; ++mt)
#pragma unroll
        for (int r = 0; r < 8; ++r) {
          float vv = acc[mt][r] + qb;
          acc[mt][r] = vv;
          atomAddLds(&red0[mt * 16 + r + rhi], vv * vv);
        }
    }
    __syncthreads();
    if (tid < TGT) rs_s[tid] = 1.f / (sqrtf(red0[tid]) + ATT_EPS);
    __syncthreads();
#pragma unroll
    for (int mt = 0; mt < 4; ++mt)
#pragma unroll
      for (int r = 0; r < 8; ++r) {
        int row = mt * 16 + r + rhi;
        sQ[row * HIDC + col] = (_Float16)(acc[mt][r] * rs_s[row]);
      }
    __syncthreads();
    { // denom = q . ks_sum + L   (4 threads per token, LDS partial reduce)
      int t  = tid >> 2;
      int c0 = (tid & 3) * 32;
      float s = 0.f;
      for (int c = c0; c < c0 + 32; ++c) s += (float)sQ[t * HIDC + c] * sKs[c];
      atomAddLds(&red1[t], s);
    }
    __syncthreads();
    if (tid < TGT) rden[tid] = 1.f / (red1[tid] + (float)L_TOK + ATT_EPS);
    __syncthreads();

    // num = q @ kvs  (M=64,K=128,N=128); fold (num+v_mean)/denom into oacc
    v8f nacc[4];
#pragma unroll
    for (int mt = 0; mt < 4; ++mt) nacc[mt] = v8zero();
    gemm_4m<4>(sQ, HIDC, kvsT + hh * HIDC * HIDC + (wv * 16) * HIDC, HIDC,
               nacc, lane);
    {
      const float vm = sVm[col];
#pragma unroll
      for (int mt = 0; mt < 4; ++mt)
#pragma unroll
        for (int r = 0; r < 8; ++r) {
          int row = mt * 16 + r + rhi;
          oacc[mt][r] += (nacc[mt][r] + vm) * rden[row];
        }
    }
  }
  __syncthreads();                         // everyone done reading sQ

  // head mean (*0.25), residual (alpha=0.5) against h
#pragma unroll
  for (int mt = 0; mt < 4; ++mt)
#pragma unroll
    for (int r = 0; r < 8; ++r) {
      int row = mt * 16 + r + rhi;
      float o = oacc[mt][r] * (0.25f * 0.5f)
              + 0.5f * (float)sH[row * HIDC + col];
      sOut[row * HIDC + col] = o;
    }
  __syncthreads();
  if (tid < TGT) {
    float s = 0.f, s2 = 0.f;
    for (int c = 0; c < HIDC; ++c) {
      float v = sOut[tid * HIDC + c];
      s += v; s2 += v * v;
    }
    float mu  = s * (1.f / HIDC);
    float var = s2 * (1.f / HIDC) - mu * mu;
    mu_s[tid] = mu;
    rs_s[tid] = rsqrtf(var + LN_EPS);
  }
  __syncthreads();
  for (int i = tid; i < TGT * HIDC; i += NTHR) {
    int t = i >> 7, c = i & 127;
    out[(size_t)tok0 * HIDC + i] =
        (sOut[i] - mu_s[t]) * rs_s[t] * ln1_g[c] + ln1_b[c];
  }
}

// ---------------------------------------------------------------------------
extern "C" void kernel_launch(void* const* d_in, const int* in_sizes, int n_in,
                              void* d_out, int out_size, void* d_ws, size_t ws_size,
                              hipStream_t stream) {
  (void)in_sizes; (void)n_in; (void)out_size; (void)ws_size;
  const float* x     = (const float*)d_in[0];
  const float* fc0_w = (const float*)d_in[1];
  const float* fc0_b = (const float*)d_in[2];
  const float* ln0_g = (const float*)d_in[3];
  const float* ln0_b = (const float*)d_in[4];
  const float* wq_w  = (const float*)d_in[5];
  const float* wq_b  = (const float*)d_in[6];
  const float* wk_w  = (const float*)d_in[7];
  const float* wk_b  = (const float*)d_in[8];
  const float* wv_w  = (const float*)d_in[9];
  const float* wv_b  = (const float*)d_in[10];
  const float* ln1_g = (const float*)d_in[11];
  const float* ln1_b = (const float*)d_in[12];
  float* out = (float*)d_out;

  // workspace carve (256 B aligned slices)
  char* p = (char*)d_ws;
  auto carve = [&](size_t bytes) {
    char* r = p; p += (bytes + 255) & ~(size_t)255; return r;
  };
  _Float16* fc0T = (_Float16*)carve(128 * 64 * sizeof(_Float16));
  _Float16* wqT  = (_Float16*)carve(512 * 128 * sizeof(_Float16));
  _Float16* wkT  = (_Float16*)carve(512 * 128 * sizeof(_Float16));
  _Float16* wvT  = (_Float16*)carve(512 * 128 * sizeof(_Float16));
  float* kvs   = (float*)carve(NHEAD * HIDC * HIDC * sizeof(float)); // contiguous
  float* ksum  = (float*)carve(NHEAD * HIDC * sizeof(float));        // with kvs
  float* vsum  = (float*)carve(NHEAD * HIDC * sizeof(float));        // and ksum
  float* vmean = (float*)carve(NHEAD * HIDC * sizeof(float));
  _Float16* kvsT = (_Float16*)carve(NHEAD * HIDC * HIDC * sizeof(_Float16));
  _Float16* hbuf = (_Float16*)carve((size_t)L_TOK * HIDC * sizeof(_Float16));

  // zero the accumulation region (kvs+ksum+vsum are contiguous: sizes are
  // multiples of 256 B so carve introduced no padding between them)
  const int nz = NHEAD * HIDC * HIDC + 2 * NHEAD * HIDC;
  k_zero<<<(nz + NTHR - 1) / NTHR, NTHR, 0, stream>>>(kvs, nz);

  // transpose+cast weights to f16 [N][K] layout
  k_cast_t<<<(64 * 128 + NTHR - 1) / NTHR, NTHR, 0, stream>>>(fc0_w, fc0T, 64, 128);
  k_cast_t<<<(128 * 512 + NTHR - 1) / NTHR, NTHR, 0, stream>>>(wq_w, wqT, 128, 512);
  k_cast_t<<<(128 * 512 + NTHR - 1) / NTHR, NTHR, 0, stream>>>(wk_w, wkT, 128, 512);
  k_cast_t<<<(128 * 512 + NTHR - 1) / NTHR, NTHR, 0, stream>>>(wv_w, wvT, 128, 512);

  k_front<<<NTILE, NTHR, 0, stream>>>(x, fc0T, fc0_b, ln0_g, ln0_b,
                                      wkT, wk_b, wvT, wv_b,
                                      hbuf, kvs, ksum, vsum);

  k_mid<<<(NHEAD * HIDC * HIDC + NTHR - 1) / NTHR, NTHR, 0, stream>>>(
      kvs, kvsT, vsum, vmean);

  k_attn<<<NTILE, NTHR, 0, stream>>>(hbuf, wqT, wq_b, kvsT, ksum, vmean,
                                     ln1_g, ln1_b, out);
}
